// EncoderBlock_10256381903252
// MI455X (gfx1250) — matrix operand order repaired and checked by reading, not compiled, not run on hardware
//
#include <hip/hip_runtime.h>
#include <math.h>

// ---------------- problem constants ----------------
#define N_EMBED 1947
#define N_HEADS 11
#define HEAD_SZ 177
#define D_FF (4 * N_EMBED)
#define BATCH 2
#define SEQ 2048
#define BS (BATCH * SEQ)
#define EPS 1e-5f

// padded (16B-aligned) leading dims for bf16 buffers
#define DP 1952             // N_EMBED -> mult of 32
#define FP 7792             // D_FF    -> mult of 16
#define EP 192              // HEAD_SZ -> mult of 32
#define HDP (N_HEADS * EP)  // 2112

typedef long long ll;

typedef __attribute__((ext_vector_type(16))) __bf16 v16bf;
typedef __attribute__((ext_vector_type(8))) __bf16 v8bf;
typedef __attribute__((ext_vector_type(4))) __bf16 v4bf;
typedef __attribute__((ext_vector_type(8))) float v8f;
typedef __attribute__((ext_vector_type(4))) float v4f;
typedef __attribute__((ext_vector_type(4))) unsigned v4u;
typedef __attribute__((ext_vector_type(8))) unsigned v8u;

// ---------------- gfx1250 Tensor Data Mover staging ----------------
// One descriptor moves a [rows=128][cols=32] bf16 tile (2D) global->LDS.
// tensor_dim0/1 are clamped to the remaining K/M so hardware zero-fills OOB
// (edge tiles need no software path). pad fields deposit rows at a 40-element
// (80B) LDS pitch: 16 DWORDs data + 4 DWORDs pad.
__device__ __forceinline__ void tdm_load_tile(void* lds, const void* gp, ll stride_elems,
                                              int rem_k, int rem_m) {
  unsigned long long ga = (unsigned long long)gp;
  unsigned td0 = rem_k < 0 ? 0u : (unsigned)rem_k;
  unsigned td1 = rem_m < 0 ? 0u : (unsigned)rem_m;
  unsigned long long st = (unsigned long long)stride_elems;
  v4u g0;
  g0[0] = 1u;                                             // count=1, user descriptor
  g0[1] = (unsigned)(unsigned long long)lds;              // lds_addr (bytes)
  g0[2] = (unsigned)ga;                                   // global_addr[31:0]
  g0[3] = (unsigned)((ga >> 32) & 0x1FFFFFFu) | (2u << 30);  // addr[56:32] | type=2
  v8u g1;
  g1[0] = (1u << 16)      // data_size = 2B
          | (1u << 20)    // pad_enable
          | (3u << 22)    // pad_interval: 16 DWORDs
          | (3u << 25);   // pad_amount: 4 DWORDs
  g1[1] = (td0 & 0xFFFFu) << 16;                             // tensor_dim0[15:0]
  g1[2] = ((td0 >> 16) & 0xFFFFu) | ((td1 & 0xFFFFu) << 16); // dim0[31:16] | dim1[15:0]
  g1[3] = ((td1 >> 16) & 0xFFFFu) | (32u << 16);             // dim1[31:16] | tile_dim0=32
  g1[4] = 128u;                                              // tile_dim1=128, tile_dim2=0
  g1[5] = (unsigned)st;                                      // dim0_stride[31:0]
  g1[6] = (unsigned)((st >> 32) & 0xFFFFu);                  // dim0_stride[47:32]
  g1[7] = 0u;
  asm volatile("tensor_load_to_lds %0, %1" ::"s"(g0), "s"(g1) : "memory");
}
__device__ __forceinline__ void wait_tensor() { __builtin_amdgcn_s_wait_tensorcnt(0); }

// ---------------- conversion / packing kernels ----------------
__global__ void k_cvt_pad(const float* __restrict__ src, __bf16* __restrict__ dst,
                          ll rows, ll cols, ll pitch) {
  ll i = (ll)blockIdx.x * blockDim.x + threadIdx.x;
  if (i >= rows * pitch) return;
  ll r = i / pitch, c = i % pitch;
  dst[i] = (c < cols) ? (__bf16)src[r * cols + c] : (__bf16)0.0f;
}

// W[h][d][e] fp32 -> dst[d][pitch] bf16, concat-head columns (c = h*177+e)
__global__ void k_pack_headw(const float* __restrict__ src, __bf16* __restrict__ dst, ll pitch) {
  ll i = (ll)blockIdx.x * blockDim.x + threadIdx.x;
  if (i >= (ll)N_EMBED * pitch) return;
  ll d = i / pitch, c = i % pitch;
  __bf16 v = (__bf16)0.0f;
  if (c < N_EMBED) {
    ll h = c / HEAD_SZ, e = c % HEAD_SZ;
    v = (__bf16)src[(h * (ll)N_EMBED + d) * HEAD_SZ + e];
  }
  dst[i] = v;
}

// ---------------- generic bf16 WMMA GEMM ----------------
#define BM 128
#define BN 128
#define BK 32
#define LP 40  // LDS row pitch in bf16 (32 data + 8 pad)

template <int A_F32>
__global__ __launch_bounds__(256) void k_gemm_wmma(
    const void* __restrict__ A, ll a_sb, ll a_sh, ll lda,
    const __bf16* __restrict__ B, ll b_sb, ll b_sh, ll bsk, ll bsn,
    float* __restrict__ Cf, __bf16* __restrict__ Cb, ll c_sb, ll c_sh, ll ldc,
    const float* __restrict__ bias, const float* __restrict__ resid, ll ldr,
    int M, int N, int K, int Hsplit, float alpha, int do_relu, int head_pad) {
  int z = blockIdx.z;
  int zb = (Hsplit > 1) ? (z / Hsplit) : z;
  int zh = (Hsplit > 1) ? (z % Hsplit) : 0;
  ll a_off = (ll)zb * a_sb + (ll)zh * a_sh;
  ll b_off = (ll)zb * b_sb + (ll)zh * b_sh;
  ll c_off = (ll)zb * c_sb + (ll)zh * c_sh;

  int tileM = blockIdx.y * BM;
  int tileN = blockIdx.x * BN;
  bool fullN = (tileN + BN <= N);

  __shared__ __align__(16) __bf16 As[2][BM][LP];
  __shared__ __align__(16) __bf16 Bs[2][BN][LP];  // transposed: [n][k]

  int tid = threadIdx.x;
  int lane = tid & 31;
  int w = tid >> 5;
  int wm = (w >> 2) * 64;  // 2 waves along M
  int wn = (w & 3) * 32;   // 4 waves along N

  const float* Af = (const float*)A;
  const __bf16* Ab = (const __bf16*)A;

  v8f acc[4][2];
#pragma unroll
  for (int i = 0; i < 4; ++i)
#pragma unroll
    for (int j = 0; j < 2; ++j) {
      v8f zv = {0.f, 0.f, 0.f, 0.f, 0.f, 0.f, 0.f, 0.f};
      acc[i][j] = zv;
    }

  auto stage = [&](int buf, int k0) {
    // ---------- TDM-staged operands (wave 0 issues; EXEC ignored by TDM) ----------
    if (w == 0) {
      if (!A_F32)
        tdm_load_tile(&As[buf][0][0], Ab + a_off + (ll)tileM * lda + k0, lda, K - k0, M - tileM);
      if (bsk == 1)  // K-contiguous B (k^T)
        tdm_load_tile(&Bs[buf][0][0], B + b_off + (ll)tileN * bsn + k0, bsn, K - k0, N - tileN);
    }
    // ---------- A fp32 (scores): vector load + convert, branchless edges ----------
    if (A_F32) {
      bool fullK = (k0 + BK <= K);
      if (fullK) {  // M is always a multiple of BM for the fp32-A GEMM
#pragma unroll
        for (int c = 0; c < 4; ++c) {
          int idx = tid + 256 * c;
          int m = idx >> 3, q = idx & 7;
          const v4f* gp = (const v4f*)(Af + a_off + (ll)(tileM + m) * lda + k0 + q * 4);
          v4f f = *gp;
          v4bf h;
          h[0] = (__bf16)f[0]; h[1] = (__bf16)f[1]; h[2] = (__bf16)f[2]; h[3] = (__bf16)f[3];
          *(v4bf*)&As[buf][m][q * 4] = h;
        }
      } else {
#pragma unroll
        for (int r = 0; r < 16; ++r) {
          int idx = tid + 256 * r;
          int m = idx >> 5, k = idx & 31;
          int gm = tileM + m, gk = k0 + k;
          bool inb = (gm < M) && (gk < K);
          float f = Af[a_off + (ll)min(gm, M - 1) * lda + min(gk, K - 1)];
          As[buf][m][k] = inb ? (__bf16)f : (__bf16)0.0f;
        }
      }
    }
    // ---------- N-contiguous B (weights / v): vector load + LDS transpose ----------
    if (bsk != 1) {
      bool fullK = (k0 + BK <= K);
      if (fullN && fullK) {
#pragma unroll
        for (int c = 0; c < 2; ++c) {
          int idx = tid + 256 * c;
          int k = idx >> 4, nq = idx & 15;
          const v8bf* gp = (const v8bf*)(B + b_off + (ll)(k0 + k) * bsk + tileN + nq * 8);
          v8bf v = *gp;
#pragma unroll
          for (int j = 0; j < 8; ++j) Bs[buf][nq * 8 + j][k] = v[j];
        }
      } else {  // branchless clamped edge path
#pragma unroll
        for (int r = 0; r < 16; ++r) {
          int idx = tid + 256 * r;
          int n = idx >> 5, k = idx & 31;
          int gn = tileN + n, gk = k0 + k;
          bool inb = (gn < N) && (gk < K);
          __bf16 v = B[b_off + (ll)min(gk, K - 1) * bsk + (ll)min(gn, N - 1) * bsn];
          Bs[buf][n][k] = inb ? v : (__bf16)0.0f;
        }
      }
    }
  };

  // fragment addressing constants (16-bit WMMA VGPR layout)
  int fm = lane & 15;
  int ksel = lane >> 4;
  int ka8 = ksel * 8;
  int kb16 = ksel * 16;

  int nk = (K + BK - 1) / BK;
  stage(0, 0);
  wait_tensor();
  __syncthreads();

  for (int t = 0; t < nk; ++t) {
    int cur = t & 1;
    if (t + 1 < nk) stage(cur ^ 1, (t + 1) * BK);  // overlap: issue next tile DMA now
    if (t + 2 < nk) {                              // L2 prefetch two tiles ahead
      ll kp = (ll)(t + 2) * BK;
      const char* pa =
          (const char*)A + (a_off + (ll)(tileM + (tid >> 1)) * lda + kp) * (A_F32 ? 4 : 2);
      __builtin_prefetch(pa, 0, 0);
      const __bf16* pb = B + b_off +
                         (bsk == 1 ? ((ll)(tileN + (tid >> 1)) * bsn + kp)
                                   : ((ll)(kp + (tid & 31)) * bsk + tileN));
      __builtin_prefetch(pb, 0, 0);
    }

    v16bf afrag[4];
    v16bf bfrag[2];
#pragma unroll
    for (int i = 0; i < 4; ++i) {
      const __bf16* ap = &As[cur][wm + 16 * i + fm][0];
      v8bf lo = *(const v8bf*)(ap + ka8);
      v8bf hi = *(const v8bf*)(ap + 16 + ka8);
      afrag[i] =
          __builtin_shufflevector(lo, hi, 0, 1, 2, 3, 4, 5, 6, 7, 8, 9, 10, 11, 12, 13, 14, 15);
    }
#pragma unroll
    for (int j = 0; j < 2; ++j) {
      const __bf16* bp = &Bs[cur][wn + 16 * j + fm][0];
      v8bf lo = *(const v8bf*)(bp + ka8);
      v8bf hi = *(const v8bf*)(bp + 16 + ka8);
      bfrag[j] =
          __builtin_shufflevector(lo, hi, 0, 1, 2, 3, 4, 5, 6, 7, 8, 9, 10, 11, 12, 13, 14, 15);
    }
#pragma unroll
    for (int i = 0; i < 4; ++i)
#pragma unroll
      for (int j = 0; j < 2; ++j)
        acc[i][j] = __builtin_amdgcn_wmma_f32_16x16x32_bf16(
            false, afrag[i], false, bfrag[j], (short)0, acc[i][j], false, false);

    wait_tensor();    // next-buffer DMA done (issued before compute -> overlapped)
    __syncthreads();  // everyone done reading cur before it is restaged
  }

  // ---------------- epilogue ----------------
  int crow0 = (lane >> 4) * 8;
  int ccol = lane & 15;
#pragma unroll
  for (int i = 0; i < 4; ++i)
#pragma unroll
    for (int j = 0; j < 2; ++j)
#pragma unroll
      for (int r = 0; r < 8; ++r) {
        int row = tileM + wm + 16 * i + crow0 + r;
        int col = tileN + wn + 16 * j + ccol;
        if (row < M && col < N) {
          float v = acc[i][j][r] * alpha;
          if (bias) v += bias[col];
          if (resid) v += resid[(ll)row * ldr + col];
          if (do_relu) v = v > 0.f ? v : 0.f;
          int colo = head_pad ? (col / HEAD_SZ) * head_pad + (col % HEAD_SZ) : col;
          ll ci = c_off + (ll)row * ldc + colo;
          if (Cf) Cf[ci] = v;
          if (Cb) Cb[ci] = (__bf16)v;
        }
      }
}

// ---------------- row softmax (in place, fp32) ----------------
__global__ __launch_bounds__(256) void k_softmax(float* __restrict__ s, int rowlen) {
  ll row = blockIdx.x;
  float* p = s + row * (ll)rowlen;
  int tid = threadIdx.x;
  __shared__ float red[8];

  float mx = -3.402823466e38f;
  for (int i = tid; i < rowlen; i += 256) mx = fmaxf(mx, p[i]);
#pragma unroll
  for (int m = 16; m; m >>= 1) mx = fmaxf(mx, __shfl_xor(mx, m, 32));
  if ((tid & 31) == 0) red[tid >> 5] = mx;
  __syncthreads();
  mx = red[0];
#pragma unroll
  for (int i = 1; i < 8; ++i) mx = fmaxf(mx, red[i]);
  __syncthreads();

  float sum = 0.f;
  for (int i = tid; i < rowlen; i += 256) {
    float e = __expf(p[i] - mx);
    p[i] = e;
    sum += e;
  }
#pragma unroll
  for (int m = 16; m; m >>= 1) sum += __shfl_xor(sum, m, 32);
  if ((tid & 31) == 0) red[tid >> 5] = sum;
  __syncthreads();
  sum = 0.f;
#pragma unroll
  for (int i = 0; i < 8; ++i) sum += red[i];
  float inv = 1.f / sum;
  for (int i = tid; i < rowlen; i += 256) p[i] *= inv;
}

// ---------------- LayerNorm (ddof=1), padded input pitch ----------------
__global__ __launch_bounds__(256) void k_layernorm(const float* __restrict__ x,
                                                   const float* __restrict__ gamma,
                                                   const float* __restrict__ beta,
                                                   float* __restrict__ out, int D, ll pitch) {
  ll row = blockIdx.x;
  const float* p = x + row * pitch;
  float* o = out + row * (ll)D;
  int tid = threadIdx.x;
  __shared__ float rs[8], rss[8];

  float s = 0.f, ss = 0.f;
  for (int i = tid; i < D; i += 256) {
    float v = p[i];
    s += v;
    ss += v * v;
  }
#pragma unroll
  for (int m = 16; m; m >>= 1) {
    s += __shfl_xor(s, m, 32);
    ss += __shfl_xor(ss, m, 32);
  }
  if ((tid & 31) == 0) {
    rs[tid >> 5] = s;
    rss[tid >> 5] = ss;
  }
  __syncthreads();
  s = 0.f;
  ss = 0.f;
#pragma unroll
  for (int i = 0; i < 8; ++i) {
    s += rs[i];
    ss += rss[i];
  }
  float mean = s / (float)D;
  float var = (ss - (float)D * mean * mean) / (float)(D - 1);
  float inv = rsqrtf(var + EPS);
  for (int i = tid; i < D; i += 256) o[i] = gamma[i] * ((p[i] - mean) * inv) + beta[i];
}

// ---------------- host launch ----------------
static inline unsigned cdiv(ll a, ll b) { return (unsigned)((a + b - 1) / b); }

extern "C" void kernel_launch(void* const* d_in, const int* in_sizes, int n_in,
                              void* d_out, int out_size, void* d_ws, size_t ws_size,
                              hipStream_t stream) {
  const float* x = (const float*)d_in[0];
  const float* Wq = (const float*)d_in[1];
  const float* Wk = (const float*)d_in[2];
  const float* Wv = (const float*)d_in[3];
  const float* bq = (const float*)d_in[4];
  const float* bk = (const float*)d_in[5];
  const float* bv = (const float*)d_in[6];
  const float* Wp = (const float*)d_in[7];
  const float* bp = (const float*)d_in[8];
  const float* W1 = (const float*)d_in[9];
  const float* b1 = (const float*)d_in[10];
  const float* W2 = (const float*)d_in[11];
  const float* b2 = (const float*)d_in[12];
  const float* gamma = (const float*)d_in[13];
  const float* beta = (const float*)d_in[14];
  float* out = (float*)d_out;

  const ll D = N_EMBED, F = D_FF, E = HEAD_SZ, H = N_HEADS, S = SEQ;

  char* ws = (char*)d_ws;
  size_t off = 0;
  auto alloc = [&](size_t bytes) -> void* {
    void* p = ws + off;
    off += (bytes + 255) & ~(size_t)255;
    return p;
  };
  __bf16* xb = (__bf16*)alloc((size_t)BS * DP * 2);
  __bf16* qb = (__bf16*)alloc((size_t)BS * HDP * 2);
  __bf16* kb = (__bf16*)alloc((size_t)BS * HDP * 2);
  __bf16* vb = (__bf16*)alloc((size_t)BS * HDP * 2);
  __bf16* ob = (__bf16*)alloc((size_t)BS * DP * 2);
  __bf16* Wqb = (__bf16*)alloc((size_t)D * DP * 2);
  __bf16* Wkb = (__bf16*)alloc((size_t)D * DP * 2);
  __bf16* Wvb = (__bf16*)alloc((size_t)D * DP * 2);
  __bf16* Wpb = (__bf16*)alloc((size_t)D * DP * 2);
  __bf16* W1b = (__bf16*)alloc((size_t)D * FP * 2);
  __bf16* W2b = (__bf16*)alloc((size_t)F * DP * 2);
  float* x1f = (float*)alloc((size_t)BS * DP * 4);
  __bf16* x1b = (__bf16*)alloc((size_t)BS * DP * 2);
  __bf16* hb = (__bf16*)alloc((size_t)BS * FP * 2);
  float* x2f = (float*)alloc((size_t)BS * DP * 4);
  float* sc = (float*)alloc((size_t)BATCH * H * S * S * 4);

  // ---- packing / conversion ----
  {
    ll n = (ll)BS * DP;
    k_cvt_pad<<<cdiv(n, 256), 256, 0, stream>>>(x, xb, BS, D, DP);
  }
  {
    ll n = D * DP;
    k_pack_headw<<<cdiv(n, 256), 256, 0, stream>>>(Wq, Wqb, DP);
    k_pack_headw<<<cdiv(n, 256), 256, 0, stream>>>(Wk, Wkb, DP);
    k_pack_headw<<<cdiv(n, 256), 256, 0, stream>>>(Wv, Wvb, DP);
    k_cvt_pad<<<cdiv(n, 256), 256, 0, stream>>>(Wp, Wpb, D, D, DP);
    n = D * FP;
    k_cvt_pad<<<cdiv(n, 256), 256, 0, stream>>>(W1, W1b, D, F, FP);
    n = F * DP;
    k_cvt_pad<<<cdiv(n, 256), 256, 0, stream>>>(W2, W2b, F, D, DP);
  }

  dim3 blk(256);
  // ---- QKV projections -> padded head layout [BS][H][EP] ----
  dim3 gD(cdiv(D, BN), cdiv(BS, BM), 1);
  k_gemm_wmma<0><<<gD, blk, 0, stream>>>(xb, 0, 0, DP, Wqb, 0, 0, DP, 1,
                                         nullptr, qb, 0, 0, HDP, bq, nullptr, 0,
                                         BS, (int)D, (int)D, 1, 1.f, 0, EP);
  k_gemm_wmma<0><<<gD, blk, 0, stream>>>(xb, 0, 0, DP, Wkb, 0, 0, DP, 1,
                                         nullptr, kb, 0, 0, HDP, bk, nullptr, 0,
                                         BS, (int)D, (int)D, 1, 1.f, 0, EP);
  k_gemm_wmma<0><<<gD, blk, 0, stream>>>(xb, 0, 0, DP, Wvb, 0, 0, DP, 1,
                                         nullptr, vb, 0, 0, HDP, bv, nullptr, 0,
                                         BS, (int)D, (int)D, 1, 1.f, 0, EP);

  // ---- scores = (q @ k^T)/sqrt(E), batched over (b,h) ----
  float alpha = 1.f / sqrtf((float)HEAD_SZ);
  dim3 gS(cdiv(S, BN), cdiv(S, BM), BATCH * N_HEADS);
  k_gemm_wmma<0><<<gS, blk, 0, stream>>>(qb, S * HDP, EP, HDP,
                                         kb, S * HDP, EP, 1, HDP,
                                         sc, nullptr, H * S * S, S * S, S,
                                         nullptr, nullptr, 0,
                                         (int)S, (int)S, (int)E, (int)H, alpha, 0, 0);

  // ---- softmax ----
  k_softmax<<<(unsigned)(BATCH * H * S), blk, 0, stream>>>(sc, (int)S);

  // ---- o = attn @ v -> unpadded concat layout [BS][DP] ----
  dim3 gO(cdiv(E, BN), cdiv(S, BM), BATCH * N_HEADS);
  k_gemm_wmma<1><<<gO, blk, 0, stream>>>(sc, H * S * S, S * S, S,
                                         vb, S * HDP, EP, HDP, 1,
                                         nullptr, ob, S * DP, E, DP,
                                         nullptr, nullptr, 0,
                                         (int)S, (int)E, (int)S, (int)H, 1.f, 0, 0);

  // ---- x1 = x + o @ Wp + bp ----
  k_gemm_wmma<0><<<gD, blk, 0, stream>>>(ob, 0, 0, DP, Wpb, 0, 0, DP, 1,
                                         x1f, x1b, 0, 0, DP, bp, x, D,
                                         BS, (int)D, (int)D, 1, 1.f, 0, 0);

  // ---- h = relu(x1 @ W1 + b1) ----
  dim3 gF(cdiv(F, BN), cdiv(BS, BM), 1);
  k_gemm_wmma<0><<<gF, blk, 0, stream>>>(x1b, 0, 0, DP, W1b, 0, 0, FP, 1,
                                         nullptr, hb, 0, 0, FP, b1, nullptr, 0,
                                         BS, (int)F, (int)D, 1, 1.f, 1, 0);

  // ---- x2 = x1 + h @ W2 + b2 ----
  k_gemm_wmma<0><<<gD, blk, 0, stream>>>(hb, 0, 0, FP, W2b, 0, 0, DP, 1,
                                         x2f, nullptr, 0, 0, DP, b2, x1f, DP,
                                         BS, (int)D, (int)F, 1, 1.f, 0, 0);

  // ---- LayerNorm ----
  k_layernorm<<<(unsigned)BS, blk, 0, stream>>>(x2f, gamma, beta, out, (int)D, (ll)DP);
}
